// Step_14044543058077
// MI455X (gfx1250) — compile-verified, matrix operand order checked by reading, not run on hardware
//
#include <hip/hip_runtime.h>
#include <hip/hip_bf16.h>

typedef __attribute__((ext_vector_type(16))) __bf16 v16bf;
typedef __attribute__((ext_vector_type(8)))  float  v8f;
typedef int vint4 __attribute__((vector_size(16)));

constexpr int Bb = 4;
constexpr int Ss = 4096;
constexpr int Dd = 1024;
constexpr int Mrows = Bb * Ss;          // 16384
constexpr float EPSV = 1e-6f;
constexpr int CH = 64;                  // chunks per sequence
constexpr int CL = 64;                  // chunk length (CH*CL == Ss)
constexpr int KSTG = 64;                // K per LDS stage
constexpr int NSTG = Dd / KSTG;         // 16 stages

__device__ __forceinline__ float sigmoidf_(float x) {
    return 1.0f / (1.0f + __expf(-x));
}

// ---- CDNA5 async global->LDS staging (ASYNCcnt path), with sync fallback ----
#if defined(__has_builtin)
#if __has_builtin(__builtin_amdgcn_global_load_async_to_lds_b128)
#define HAS_ASYNC_LDS 1
#endif
#endif

#if defined(HAS_ASYNC_LDS)
#define ASYNC_COPY16(lp, gp)                                                  \
    __builtin_amdgcn_global_load_async_to_lds_b128(                           \
        (__attribute__((address_space(1))) vint4*)(gp),                       \
        (__attribute__((address_space(3))) vint4*)(lp), 0, 0)
#if __has_builtin(__builtin_amdgcn_s_wait_asynccnt)
#define WAIT_ASYNC(n) __builtin_amdgcn_s_wait_asynccnt(n)
#else
#define WAIT_ASYNC(n) asm volatile("s_wait_asynccnt %0" ::"i"(n) : "memory")
#endif
#else
#define ASYNC_COPY16(lp, gp) (*(int4*)(lp) = *(const int4*)(gp))
#define WAIT_ASYNC(n)
#endif

// ---------------------------------------------------------------------------
// K0: transpose f32 [K][N] weight -> bf16 [N][K] (B fragments contiguous).
// ---------------------------------------------------------------------------
__global__ __launch_bounds__(256) void transpose_bf16_kernel(const float* __restrict__ src,
                                                             __hip_bfloat16* __restrict__ dst) {
    __shared__ float tile[32][33];
    int tx = threadIdx.x, ty = threadIdx.y;   // 32 x 8
    int n = blockIdx.x * 32 + tx;
#pragma unroll
    for (int i = 0; i < 4; ++i) {
        int k = blockIdx.y * 32 + ty + i * 8;
        tile[ty + i * 8][tx] = src[(size_t)k * Dd + n];
    }
    __syncthreads();
#pragma unroll
    for (int i = 0; i < 4; ++i) {
        int nn = blockIdx.x * 32 + ty + i * 8;
        int kk = blockIdx.y * 32 + tx;
        dst[(size_t)nn * Dd + kk] = __float2bfloat16(tile[tx][ty + i * 8]);
    }
}

// ---------------------------------------------------------------------------
// K1: gate = rmsnorm(c, scale_c) + rmsnorm(ctx, scale_ctx), stored bf16.
// ---------------------------------------------------------------------------
__global__ __launch_bounds__(256) void gate_kernel(const float* __restrict__ c,
                                                   const float* __restrict__ ctx,
                                                   const float* __restrict__ sc_c,
                                                   const float* __restrict__ sc_x,
                                                   __hip_bfloat16* __restrict__ gate) {
    int row = blockIdx.x;
    const float* cr = c + (size_t)row * Dd;
    const float* xr = ctx + (size_t)row * Dd;
    float cv[4], xv[4];
    float s0 = 0.f, s1 = 0.f;
#pragma unroll
    for (int i = 0; i < 4; ++i) {
        int d = threadIdx.x + i * 256;
        cv[i] = cr[d]; xv[i] = xr[d];
        s0 = fmaf(cv[i], cv[i], s0);
        s1 = fmaf(xv[i], xv[i], s1);
    }
    for (int off = 16; off > 0; off >>= 1) {
        s0 += __shfl_down(s0, off, 32);
        s1 += __shfl_down(s1, off, 32);
    }
    __shared__ float r0[8], r1[8];
    int wv = threadIdx.x >> 5, ln = threadIdx.x & 31;
    if (ln == 0) { r0[wv] = s0; r1[wv] = s1; }
    __syncthreads();
    if (threadIdx.x == 0) {
        float t0 = 0.f, t1 = 0.f;
        for (int i = 0; i < 8; ++i) { t0 += r0[i]; t1 += r1[i]; }
        r0[0] = rsqrtf(t0 * (1.0f / Dd) + EPSV);
        r1[0] = rsqrtf(t1 * (1.0f / Dd) + EPSV);
    }
    __syncthreads();
    float rc = r0[0], rx = r1[0];
    __hip_bfloat16* gr = gate + (size_t)row * Dd;
#pragma unroll
    for (int i = 0; i < 4; ++i) {
        int d = threadIdx.x + i * 256;
        float g = cv[i] * rc * sc_c[d] + xv[i] * rx * sc_x[d];
        gr[d] = __float2bfloat16(g);
    }
}

// ---------------------------------------------------------------------------
// K2: dual GEMM (gate@wT_alpha, gate@wT_beta), bf16 WMMA.
// Block = 8 waves, tile M=128 x N=64. Weight panels staged to LDS in
// fragment-major layout via async-to-LDS, double-buffered over K stages.
// Fused epilogue -> alpha, x.
// ---------------------------------------------------------------------------
__global__ __launch_bounds__(256) void dual_gemm_kernel(const __hip_bfloat16* __restrict__ A,
                                                        const __hip_bfloat16* __restrict__ WA,
                                                        const __hip_bfloat16* __restrict__ WB,
                                                        const float* __restrict__ bA,
                                                        const float* __restrict__ bB,
                                                        const float* __restrict__ v,
                                                        float* __restrict__ alpha,
                                                        float* __restrict__ xo) {
    // [buf][weight][kblock(32)][nt][lane][16 halves]  == fragment-major, 32KB
    __shared__ __bf16 smem[2][2][2][4][32][16];

    int tid  = threadIdx.x;
    int lane = tid & 31;
    int wave = tid >> 5;
    int m0 = blockIdx.y * 128 + wave * 16;
    int n0 = blockIdx.x * 64;
    int l15 = lane & 15;
    int kb  = (lane >> 4) * 16;

    // issue one stage's weight panels (4 async b128 per thread)
    auto issue = [&](int buf, int ks) {
#pragma unroll
        for (int w = 0; w < 2; ++w) {
            const __hip_bfloat16* Wp = w ? WB : WA;
#pragma unroll
            for (int r = 0; r < 2; ++r) {
                int cId = tid + r * 256;         // 512 chunks of 16B per weight
                int n   = cId >> 3;              // 0..63
                int k0  = (cId & 7) * 8;         // 0..56 (elems)
                int kblk = k0 >> 5;
                int krem = k0 & 31;
                int lnn  = ((krem >> 4) << 4) + (n & 15);
                int eo   = krem & 8;             // element offset 0 or 8
                int nt   = n >> 4;
                const __hip_bfloat16* gp = Wp + (size_t)(n0 + n) * Dd + ks * KSTG + k0;
                __bf16* lp = &smem[buf][w][kblk][nt][lnn][eo];
                ASYNC_COPY16(lp, gp);
            }
        }
    };

    const v8f zero = {0.f, 0.f, 0.f, 0.f, 0.f, 0.f, 0.f, 0.f};
    v8f acc1[4] = {zero, zero, zero, zero};
    v8f acc2[4] = {zero, zero, zero, zero};

    const __hip_bfloat16* ap = A + (size_t)(m0 + l15) * Dd + kb;

    issue(0, 0);
    v16bf a0 = *reinterpret_cast<const v16bf*>(ap);
    v16bf a1 = *reinterpret_cast<const v16bf*>(ap + 32);

    for (int ks = 0; ks < NSTG; ++ks) {
        v16bf an0 = a0, an1 = a1;
        if (ks + 1 < NSTG) {
            issue((ks + 1) & 1, ks + 1);
            an0 = *reinterpret_cast<const v16bf*>(ap + (ks + 1) * KSTG);
            an1 = *reinterpret_cast<const v16bf*>(ap + (ks + 1) * KSTG + 32);
            WAIT_ASYNC(4);   // previous stage's 4 async loads (per wave) done
        } else {
            WAIT_ASYNC(0);
        }
        __syncthreads();
        int buf = ks & 1;
#pragma unroll
        for (int kk = 0; kk < 2; ++kk) {
            v16bf af = kk ? a1 : a0;
#pragma unroll
            for (int nt = 0; nt < 4; ++nt) {
                v16bf b1 = *reinterpret_cast<const v16bf*>(&smem[buf][0][kk][nt][lane][0]);
                acc1[nt] = __builtin_amdgcn_wmma_f32_16x16x32_bf16(
                    false, af, false, b1, (short)0, acc1[nt], false, false);
                v16bf b2 = *reinterpret_cast<const v16bf*>(&smem[buf][1][kk][nt][lane][0]);
                acc2[nt] = __builtin_amdgcn_wmma_f32_16x16x32_bf16(
                    false, af, false, b2, (short)0, acc2[nt], false, false);
            }
        }
        __syncthreads();
        a0 = an0; a1 = an1;
    }

    int mbase = m0 + (lane >> 4) * 8;
#pragma unroll
    for (int nt = 0; nt < 4; ++nt) {
        int n = n0 + nt * 16 + l15;
        float ba = bA[n], bb = bB[n];
#pragma unroll
        for (int j = 0; j < 8; ++j) {
            int m = mbase + j;
            float a  = sigmoidf_(acc1[nt][j] + ba);
            float g2 = acc2[nt][j] + bb;
            float be = g2 * sigmoidf_(g2);
            float vv = v[(size_t)m * Dd + n];
            float ws = sqrtf(fmaxf(1.0f - a * a, 1e-6f));
            alpha[(size_t)m * Dd + n] = a;
            xo[(size_t)m * Dd + n]    = vv * be * ws;
        }
    }
}

// ---------------------------------------------------------------------------
// K6: single GEMM silu(fetched@wT_upd + b_upd) -> U. Same staging scheme.
// ---------------------------------------------------------------------------
__global__ __launch_bounds__(256) void upd_gemm_kernel(const __hip_bfloat16* __restrict__ A,
                                                       const __hip_bfloat16* __restrict__ W,
                                                       const float* __restrict__ bU,
                                                       float* __restrict__ U) {
    __shared__ __bf16 smem[2][2][4][32][16];   // [buf][kblock][nt][lane][16], 16KB

    int tid  = threadIdx.x;
    int lane = tid & 31;
    int wave = tid >> 5;
    int m0 = blockIdx.y * 128 + wave * 16;
    int n0 = blockIdx.x * 64;
    int l15 = lane & 15;
    int kb  = (lane >> 4) * 16;

    auto issue = [&](int buf, int ks) {
#pragma unroll
        for (int r = 0; r < 2; ++r) {
            int cId = tid + r * 256;
            int n   = cId >> 3;
            int k0  = (cId & 7) * 8;
            int kblk = k0 >> 5;
            int krem = k0 & 31;
            int lnn  = ((krem >> 4) << 4) + (n & 15);
            int eo   = krem & 8;
            int nt   = n >> 4;
            const __hip_bfloat16* gp = W + (size_t)(n0 + n) * Dd + ks * KSTG + k0;
            __bf16* lp = &smem[buf][kblk][nt][lnn][eo];
            ASYNC_COPY16(lp, gp);
        }
    };

    const v8f zero = {0.f, 0.f, 0.f, 0.f, 0.f, 0.f, 0.f, 0.f};
    v8f acc[4] = {zero, zero, zero, zero};

    const __hip_bfloat16* ap = A + (size_t)(m0 + l15) * Dd + kb;

    issue(0, 0);
    v16bf a0 = *reinterpret_cast<const v16bf*>(ap);
    v16bf a1 = *reinterpret_cast<const v16bf*>(ap + 32);

    for (int ks = 0; ks < NSTG; ++ks) {
        v16bf an0 = a0, an1 = a1;
        if (ks + 1 < NSTG) {
            issue((ks + 1) & 1, ks + 1);
            an0 = *reinterpret_cast<const v16bf*>(ap + (ks + 1) * KSTG);
            an1 = *reinterpret_cast<const v16bf*>(ap + (ks + 1) * KSTG + 32);
            WAIT_ASYNC(2);
        } else {
            WAIT_ASYNC(0);
        }
        __syncthreads();
        int buf = ks & 1;
#pragma unroll
        for (int kk = 0; kk < 2; ++kk) {
            v16bf af = kk ? a1 : a0;
#pragma unroll
            for (int nt = 0; nt < 4; ++nt) {
                v16bf bf = *reinterpret_cast<const v16bf*>(&smem[buf][kk][nt][lane][0]);
                acc[nt] = __builtin_amdgcn_wmma_f32_16x16x32_bf16(
                    false, af, false, bf, (short)0, acc[nt], false, false);
            }
        }
        __syncthreads();
        a0 = an0; a1 = an1;
    }

    int mbase = m0 + (lane >> 4) * 8;
#pragma unroll
    for (int nt = 0; nt < 4; ++nt) {
        int n = n0 + nt * 16 + l15;
        float b = bU[n];
#pragma unroll
        for (int j = 0; j < 8; ++j) {
            int m = mbase + j;
            float u = acc[nt][j] + b;
            U[(size_t)m * Dd + n] = u * sigmoidf_(u);
        }
    }
}

// ---------------------------------------------------------------------------
// Scan phase 1: per (b, d, chunk) compute (prod alpha, local h) over CL steps.
// ---------------------------------------------------------------------------
__global__ __launch_bounds__(256) void scan1_kernel(const float* __restrict__ alpha,
                                                    const float* __restrict__ x,
                                                    float* __restrict__ cA,
                                                    float* __restrict__ cHh) {
    int gid = blockIdx.x * 256 + threadIdx.x;    // B*CH*Dd threads
    int d = gid & (Dd - 1);
    int rest = gid >> 10;                        // b*CH + chunk
    int b = rest >> 6;
    int chunk = rest & (CH - 1);
    size_t base = ((size_t)(b * Ss + chunk * CL)) * Dd + d;
    float A = 1.f, H = 0.f;
    for (int i = 0; i < CL; ++i) {
        size_t idx = base + (size_t)i * Dd;
        float a = alpha[idx];
        H = fmaf(a, H, x[idx]);
        A *= a;
    }
    size_t cidx = (size_t)rest * Dd + d;
    cA[cidx] = A;
    cHh[cidx] = H;
}

// Scan phase 2: sequential combine across chunks -> exclusive carry-in.
__global__ __launch_bounds__(256) void scan2_kernel(const float* __restrict__ cA,
                                                    const float* __restrict__ cHh,
                                                    float* __restrict__ cin) {
    int gid = blockIdx.x * 256 + threadIdx.x;    // B*Dd threads
    int d = gid & (Dd - 1);
    int b = gid >> 10;
    float h = 0.f;
    for (int ch = 0; ch < CH; ++ch) {
        size_t idx = (size_t)(b * CH + ch) * Dd + d;
        cin[idx] = h;
        h = fmaf(cA[idx], h, cHh[idx]);
    }
}

// Scan phase 3: replay with carry-in; emit fetched (bf16) and out+fetched.
__global__ __launch_bounds__(256) void scan3_kernel(const float* __restrict__ alpha,
                                                    const float* __restrict__ x,
                                                    const float* __restrict__ cin,
                                                    const float* __restrict__ out_in,
                                                    float* __restrict__ out2,
                                                    __hip_bfloat16* __restrict__ fB) {
    int gid = blockIdx.x * 256 + threadIdx.x;
    int d = gid & (Dd - 1);
    int rest = gid >> 10;
    int b = rest >> 6;
    int chunk = rest & (CH - 1);
    size_t base = ((size_t)(b * Ss + chunk * CL)) * Dd + d;
    float h = cin[(size_t)rest * Dd + d];
    for (int i = 0; i < CL; ++i) {
        size_t idx = base + (size_t)i * Dd;
        h = fmaf(alpha[idx], h, x[idx]);
        fB[idx] = __float2bfloat16(h);
        out2[idx] = out_in[idx] + h;
    }
}

// ---------------------------------------------------------------------------
// K7: update = rmsnorm(U, scale_upd); d_out section1 = ctx + update.
// ---------------------------------------------------------------------------
__global__ __launch_bounds__(256) void final_kernel(const float* __restrict__ U,
                                                    const float* __restrict__ ctx,
                                                    const float* __restrict__ sc,
                                                    float* __restrict__ out1) {
    int row = blockIdx.x;
    const float* ur = U + (size_t)row * Dd;
    float uv[4];
    float s = 0.f;
#pragma unroll
    for (int i = 0; i < 4; ++i) {
        int d = threadIdx.x + i * 256;
        uv[i] = ur[d];
        s = fmaf(uv[i], uv[i], s);
    }
    for (int off = 16; off > 0; off >>= 1) s += __shfl_down(s, off, 32);
    __shared__ float r0[8];
    int wv = threadIdx.x >> 5, ln = threadIdx.x & 31;
    if (ln == 0) r0[wv] = s;
    __syncthreads();
    if (threadIdx.x == 0) {
        float t = 0.f;
        for (int i = 0; i < 8; ++i) t += r0[i];
        r0[0] = rsqrtf(t * (1.0f / Dd) + EPSV);
    }
    __syncthreads();
    float r = r0[0];
    const float* cr = ctx + (size_t)row * Dd;
    float* orow = out1 + (size_t)row * Dd;
#pragma unroll
    for (int i = 0; i < 4; ++i) {
        int d = threadIdx.x + i * 256;
        orow[d] = cr[d] + uv[i] * r * sc[d];
    }
}

// ---------------------------------------------------------------------------
extern "C" void kernel_launch(void* const* d_in, const int* in_sizes, int n_in,
                              void* d_out, int out_size, void* d_ws, size_t ws_size,
                              hipStream_t stream) {
    const float* v       = (const float*)d_in[0];
    const float* ctx     = (const float*)d_in[1];
    const float* out_in  = (const float*)d_in[2];
    const float* c       = (const float*)d_in[3];
    const float* w_alpha = (const float*)d_in[4];
    const float* b_alpha = (const float*)d_in[5];
    const float* w_beta  = (const float*)d_in[6];
    const float* b_beta  = (const float*)d_in[7];
    const float* w_upd   = (const float*)d_in[8];
    const float* b_upd   = (const float*)d_in[9];
    const float* scale_c   = (const float*)d_in[10];
    const float* scale_ctx = (const float*)d_in[11];
    const float* scale_upd = (const float*)d_in[12];

    const size_t NE = (size_t)Mrows * Dd;
    float* o0 = (float*)d_out;                   // v (pass-through)
    float* o1 = o0 + NE;                         // ctx + update
    float* o2 = o1 + NE;                         // out + fetched
    float* o3 = o2 + NE;                         // c (pass-through)

    // ----- workspace carve-up -----
    char* p = (char*)d_ws;
    __hip_bfloat16* gateB = (__hip_bfloat16*)p;  p += NE * 2;
    __hip_bfloat16* wTa   = (__hip_bfloat16*)p;  p += (size_t)Dd * Dd * 2;
    __hip_bfloat16* wTb   = (__hip_bfloat16*)p;  p += (size_t)Dd * Dd * 2;
    __hip_bfloat16* wTu   = (__hip_bfloat16*)p;  p += (size_t)Dd * Dd * 2;
    float* alpha  = (float*)p;                   p += NE * 4;
    float* xbuf   = (float*)p;                   p += NE * 4;   // reused as U
    __hip_bfloat16* fetchB = (__hip_bfloat16*)p; p += NE * 2;
    float* cA  = (float*)p;                      p += (size_t)Bb * CH * Dd * 4;
    float* cHh = (float*)p;                      p += (size_t)Bb * CH * Dd * 4;
    float* cin = (float*)p;                      p += (size_t)Bb * CH * Dd * 4;

    (void)hipMemcpyAsync(o0, v, NE * sizeof(float), hipMemcpyDeviceToDevice, stream);
    (void)hipMemcpyAsync(o3, c, NE * sizeof(float), hipMemcpyDeviceToDevice, stream);

    dim3 tb(32, 8), tg(Dd / 32, Dd / 32);
    transpose_bf16_kernel<<<tg, tb, 0, stream>>>(w_alpha, wTa);
    transpose_bf16_kernel<<<tg, tb, 0, stream>>>(w_beta,  wTb);
    transpose_bf16_kernel<<<tg, tb, 0, stream>>>(w_upd,   wTu);

    gate_kernel<<<Mrows, 256, 0, stream>>>(c, ctx, scale_c, scale_ctx, gateB);

    dim3 gg(Dd / 64, Mrows / 128);
    dual_gemm_kernel<<<gg, 256, 0, stream>>>(gateB, wTa, wTb, b_alpha, b_beta,
                                             v, alpha, xbuf);

    scan1_kernel<<<(Bb * CH * Dd) / 256, 256, 0, stream>>>(alpha, xbuf, cA, cHh);
    scan2_kernel<<<(Bb * Dd) / 256, 256, 0, stream>>>(cA, cHh, cin);
    scan3_kernel<<<(Bb * CH * Dd) / 256, 256, 0, stream>>>(alpha, xbuf, cin,
                                                           out_in, o2, fetchB);

    upd_gemm_kernel<<<gg, 256, 0, stream>>>(fetchB, wTu, b_upd, xbuf);

    final_kernel<<<Mrows, 256, 0, stream>>>(xbuf, ctx, scale_upd, o1);
}
